// ANodeGCN_309237645639
// MI455X (gfx1250) — matrix operand order repaired and checked by reading, not compiled, not run on hardware
//
#include <hip/hip_runtime.h>

typedef __bf16 bf16_t;
typedef __attribute__((ext_vector_type(8)))  bf16_t v8bf;
typedef __attribute__((ext_vector_type(16))) bf16_t v16bf;
typedef __attribute__((ext_vector_type(8)))  float  v8f;
typedef __attribute__((ext_vector_type(4)))  int    v4i;

#define NN   1024
#define BB   128
#define DIN  64
#define DOUT 64
#define EEMB 16
#define JJ   (BB * DIN)   // 8192
#define BSTR 40           // padded LDS column stride (elements): conflict-free, 16B chunks

// gfx1250 async global->LDS staging (ASYNCcnt path), guarded so the compile
// falls back to the VGPR round-trip path if the builtin is absent.
#if __has_builtin(__builtin_amdgcn_global_load_async_to_lds_b128)
#define USE_ASYNC_LDS 1
typedef __attribute__((address_space(1))) v4i gv4i;   // global (AS1) int4
typedef __attribute__((address_space(3))) v4i lv4i;   // LDS    (AS3) int4
#else
#define USE_ASYNC_LDS 0
#endif

#if __has_builtin(__builtin_amdgcn_s_wait_asynccnt)
#define WAIT_ASYNC0() __builtin_amdgcn_s_wait_asynccnt(0)
#else
#define WAIT_ASYNC0() asm volatile("s_wait_asynccnt 0x0" ::: "memory")
#endif

// ---------------------------------------------------------------------------
// Kernel 1: supports[n][m] = softmax_m( relu(emb[n] . emb[m]) ), stored bf16.
// ---------------------------------------------------------------------------
__global__ __launch_bounds__(256) void k_supports(const float* __restrict__ emb,
                                                  bf16_t* __restrict__ sup) {
    const int n = blockIdx.x;
    const int t = threadIdx.x;
    __shared__ float er[EEMB];
    __shared__ float red[256];
    if (t < EEMB) er[t] = emb[n * EEMB + t];
    __syncthreads();
    float e[EEMB];
#pragma unroll
    for (int d = 0; d < EEMB; ++d) e[d] = er[d];

    float s[4];
    float mx = -3.0e38f;
#pragma unroll
    for (int p = 0; p < 4; ++p) {
        const int m = t + p * 256;
        const float4* q = (const float4*)(emb + m * EEMB);
        float4 q0 = q[0], q1 = q[1], q2 = q[2], q3 = q[3];
        float d = e[0]*q0.x + e[1]*q0.y + e[2]*q0.z  + e[3]*q0.w
                + e[4]*q1.x + e[5]*q1.y + e[6]*q1.z  + e[7]*q1.w
                + e[8]*q2.x + e[9]*q2.y + e[10]*q2.z + e[11]*q2.w
                + e[12]*q3.x+ e[13]*q3.y+ e[14]*q3.z + e[15]*q3.w;
        d = fmaxf(d, 0.0f);
        s[p] = d;
        mx = fmaxf(mx, d);
    }
    red[t] = mx; __syncthreads();
    for (int st = 128; st > 0; st >>= 1) {
        if (t < st) red[t] = fmaxf(red[t], red[t + st]);
        __syncthreads();
    }
    mx = red[0]; __syncthreads();

    float sm = 0.f;
#pragma unroll
    for (int p = 0; p < 4; ++p) { s[p] = __expf(s[p] - mx); sm += s[p]; }
    red[t] = sm; __syncthreads();
    for (int st = 128; st > 0; st >>= 1) {
        if (t < st) red[t] += red[t + st];
        __syncthreads();
    }
    const float inv = 1.0f / red[0];
#pragma unroll
    for (int p = 0; p < 4; ++p)
        sup[(size_t)n * NN + t + p * 256] = (bf16_t)(s[p] * inv);
}

// ---------------------------------------------------------------------------
// Kernel 2: xtT[b*64+c][m] = bf16(x[b][m][c])  (LDS-tiled transpose, padded)
// ---------------------------------------------------------------------------
__global__ __launch_bounds__(256) void k_xpose(const float* __restrict__ x,
                                               bf16_t* __restrict__ xtT) {
    const int b  = blockIdx.x >> 4;
    const int m0 = (blockIdx.x & 15) * 64;
    const int t  = threadIdx.x;
    __shared__ float tile[64][65];
    const float* src = x + ((size_t)b * NN + m0) * DIN;
#pragma unroll
    for (int p = 0; p < 16; ++p) {
        const int idx = t + p * 256;
        tile[idx >> 6][idx & 63] = src[idx];
    }
    __syncthreads();
    const int c  = t >> 2;
    const int kq = (t & 3) * 16;
    v16bf v;
#pragma unroll
    for (int i = 0; i < 16; ++i) v[i] = (bf16_t)tile[kq + i][c];
    *(v16bf*)(xtT + ((size_t)(b * DIN + c)) * NN + m0 + kq) = v;
}

// ---------------------------------------------------------------------------
// Kernel 3: wT[n][o][i] = sum_d emb[n][d] * wpool[d][i][o]  (bf16, transposed)
//           bias[n][o]  = sum_d emb[n][d] * bpool[d][o]     (f32)
// ---------------------------------------------------------------------------
__global__ __launch_bounds__(256) void k_weights(const float* __restrict__ emb,
                                                 const float* __restrict__ wp,
                                                 const float* __restrict__ bp,
                                                 bf16_t* __restrict__ wT,
                                                 float* __restrict__ bias) {
    const int n = blockIdx.x;
    const int t = threadIdx.x;
    __shared__ float er[EEMB];
    if (t < EEMB) er[t] = emb[n * EEMB + t];
    __syncthreads();
    const int o  = t >> 2;
    const int ib = (t & 3) * 16;
    float acc[16];
#pragma unroll
    for (int i = 0; i < 16; ++i) acc[i] = 0.f;
    for (int d = 0; d < EEMB; ++d) {
        const float ed = er[d];
        const float* q = wp + d * (DIN * DOUT) + o;
#pragma unroll
        for (int i = 0; i < 16; ++i) acc[i] += ed * q[(ib + i) * DOUT];
    }
    v16bf v;
#pragma unroll
    for (int i = 0; i < 16; ++i) v[i] = (bf16_t)acc[i];
    *(v16bf*)(wT + (size_t)n * (DIN * DOUT) + o * DIN + ib) = v;

    if (t < DOUT) {
        float a = 0.f;
#pragma unroll
        for (int d = 0; d < EEMB; ++d) a += er[d] * bp[d * DOUT + t];
        bias[n * DOUT + t] = a;
    }
}

// ---------------------------------------------------------------------------
// Fragment helpers
// ---------------------------------------------------------------------------
__device__ __forceinline__ v16bf load_afrag(const bf16_t* __restrict__ arow, int k0) {
    v8bf lo = *(const v8bf*)(arow + k0);
    v8bf hi = *(const v8bf*)(arow + k0 + 16);
    return __builtin_shufflevector(lo, hi, 0,1,2,3,4,5,6,7,8,9,10,11,12,13,14,15);
}
__device__ __forceinline__ v16bf frag16(const bf16_t* p) {
    v8bf lo = *(const v8bf*)(p);
    v8bf hi = *(const v8bf*)(p + 16);
    return __builtin_shufflevector(lo, hi, 0,1,2,3,4,5,6,7,8,9,10,11,12,13,14,15);
}

// ---------------------------------------------------------------------------
// Kernel 4 (dominant GEMM): xg[n][j] = sum_m sup[n][m] * xtT[j][m]
// [1024 x 8192], K = 1024. Workgroup tile = 128x64 (8 waves x 16 rows), all
// waves share one 64-col B strip staged per K-step into double-buffered LDS.
// Staging uses GLOBAL_LOAD_ASYNC_TO_LDS_B128 (ASYNCcnt) when available, else
// a pipelined VGPR round-trip. Rolled loop (unroll 1) keeps accumulators in
// fixed registers: no v_mov_b64 shuffles, no WMMA->VALU hazard NOPs.
// ---------------------------------------------------------------------------
__global__ __launch_bounds__(256) void k_gemm1(const bf16_t* __restrict__ sup,
                                               const bf16_t* __restrict__ xtT,
                                               bf16_t* __restrict__ xg) {
    const int t     = threadIdx.x;
    const int lane  = t & 31;
    const int wv    = t >> 5;
    const int lhalf = lane >> 4;
    const int l15   = lane & 15;
    const int n0    = (blockIdx.x >> 7) * 128;   // 8 M-blocks
    const int j0    = (blockIdx.x & 127) * 64;   // 128 j-strips

    __shared__ __align__(128) bf16_t ldsB[2][64 * BSTR];  // 2 x 5 KB

    // staging map: thread -> (column, 16B chunk of 32 K-values)
    const int scol = t >> 2;
    const int sq   = (t & 3) * 8;
    const bf16_t* sgp = xtT + (size_t)(j0 + scol) * NN + sq;
    const int sofs = scol * BSTR + sq;

    // per-wave A row (16 private rows per wave)
    const bf16_t* arow = sup + (size_t)(n0 + wv * 16 + l15) * NN + lhalf * 8;
    // per-lane B fragment base inside an LDS buffer
    const int bofs = l15 * BSTR + lhalf * 8;

    v8f acc0 = {}, acc1 = {}, acc2 = {}, acc3 = {};

    // prologue: stage K-step 0 into buffer 0
#if USE_ASYNC_LDS
    __builtin_amdgcn_global_load_async_to_lds_b128(
        (gv4i*)(sgp), (lv4i*)&ldsB[0][sofs], 0, 0);
    WAIT_ASYNC0();
#else
    {
        v8bf g = *(const v8bf*)(sgp);
        *(v8bf*)&ldsB[0][sofs] = g;
    }
#endif
    __syncthreads();

#pragma unroll 1
    for (int s = 0; s < 31; ++s) {
        const int k0  = s * 32;
        const int cur = s & 1;
        const int nxt = cur ^ 1;

        // issue staging for K-step s+1 early (overlaps with WMMAs below);
        // buffer 'nxt' reads finished at the previous barrier.
#if USE_ASYNC_LDS
        __builtin_amdgcn_global_load_async_to_lds_b128(
            (gv4i*)(sgp + k0 + 32), (lv4i*)&ldsB[nxt][sofs], 0, 0);
#else
        v8bf g = *(const v8bf*)(sgp + k0 + 32);
#endif

        // A fragment (direct global) + B fragments (LDS)
        v16bf a  = load_afrag(arow, k0);
        const bf16_t* bb = &ldsB[cur][bofs];
        v16bf b0 = frag16(bb + 0 * 16 * BSTR);
        v16bf b1 = frag16(bb + 1 * 16 * BSTR);
        v16bf b2 = frag16(bb + 2 * 16 * BSTR);
        v16bf b3 = frag16(bb + 3 * 16 * BSTR);

        acc0 = __builtin_amdgcn_wmma_f32_16x16x32_bf16(false, a, false, b0, (short)0, acc0, false, false);
        acc1 = __builtin_amdgcn_wmma_f32_16x16x32_bf16(false, a, false, b1, (short)0, acc1, false, false);
        acc2 = __builtin_amdgcn_wmma_f32_16x16x32_bf16(false, a, false, b2, (short)0, acc2, false, false);
        acc3 = __builtin_amdgcn_wmma_f32_16x16x32_bf16(false, a, false, b3, (short)0, acc3, false, false);

#if USE_ASYNC_LDS
        WAIT_ASYNC0();                // staged tile fully in LDS
#else
        *(v8bf*)&ldsB[nxt][sofs] = g; // commit staged tile
#endif
        __syncthreads();
    }

    // epilogue: K-step 31 from buffer 1 (staged at s == 30)
    {
        const int k0 = 31 * 32;
        v16bf a  = load_afrag(arow, k0);
        const bf16_t* bb = &ldsB[1][bofs];
        v16bf b0 = frag16(bb + 0 * 16 * BSTR);
        v16bf b1 = frag16(bb + 1 * 16 * BSTR);
        v16bf b2 = frag16(bb + 2 * 16 * BSTR);
        v16bf b3 = frag16(bb + 3 * 16 * BSTR);
        acc0 = __builtin_amdgcn_wmma_f32_16x16x32_bf16(false, a, false, b0, (short)0, acc0, false, false);
        acc1 = __builtin_amdgcn_wmma_f32_16x16x32_bf16(false, a, false, b1, (short)0, acc1, false, false);
        acc2 = __builtin_amdgcn_wmma_f32_16x16x32_bf16(false, a, false, b2, (short)0, acc2, false, false);
        acc3 = __builtin_amdgcn_wmma_f32_16x16x32_bf16(false, a, false, b3, (short)0, acc3, false, false);
    }

    const int rbase = n0 + wv * 16 + lhalf * 8;
#pragma unroll
    for (int r = 0; r < 8; ++r) {
        bf16_t* dst = xg + (size_t)(rbase + r) * JJ + j0 + l15;
        dst[0]  = (bf16_t)acc0[r];
        dst[16] = (bf16_t)acc1[r];
        dst[32] = (bf16_t)acc2[r];
        dst[48] = (bf16_t)acc3[r];
    }
}

// ---------------------------------------------------------------------------
// Kernel 5: out[b][n][o] = sum_i xg[n][b*64+i] * wT[n][o][i] + bias[n][o]
// ---------------------------------------------------------------------------
__global__ __launch_bounds__(256) void k_gemm2(const bf16_t* __restrict__ xg,
                                               const bf16_t* __restrict__ wT,
                                               const float* __restrict__ bias,
                                               float* __restrict__ out) {
    const int n     = blockIdx.x;
    const int wv    = threadIdx.x >> 5;
    const int lane  = threadIdx.x & 31;
    const int lhalf = lane >> 4;
    const int l15   = lane & 15;
    const int b0    = wv * 16;

    v8f acc0, acc1, acc2, acc3;
    {
        const float* bs = bias + n * DOUT + l15;
        const float v0 = bs[0], v1 = bs[16], v2 = bs[32], v3 = bs[48];
#pragma unroll
        for (int r = 0; r < 8; ++r) { acc0[r] = v0; acc1[r] = v1; acc2[r] = v2; acc3[r] = v3; }
    }

    const bf16_t* abase = xg + (size_t)n * JJ + (b0 + l15) * DIN + lhalf * 8;
    const bf16_t* wbase = wT + (size_t)n * (DIN * DOUT) + l15 * DIN + lhalf * 8;

#pragma unroll
    for (int k0 = 0; k0 < DIN; k0 += 32) {
        v16bf a = load_afrag(abase, k0);
        v16bf bw0 = frag16(wbase + 0 * 16 * DIN + k0);
        v16bf bw1 = frag16(wbase + 1 * 16 * DIN + k0);
        v16bf bw2 = frag16(wbase + 2 * 16 * DIN + k0);
        v16bf bw3 = frag16(wbase + 3 * 16 * DIN + k0);
        acc0 = __builtin_amdgcn_wmma_f32_16x16x32_bf16(false, a, false, bw0, (short)0, acc0, false, false);
        acc1 = __builtin_amdgcn_wmma_f32_16x16x32_bf16(false, a, false, bw1, (short)0, acc1, false, false);
        acc2 = __builtin_amdgcn_wmma_f32_16x16x32_bf16(false, a, false, bw2, (short)0, acc2, false, false);
        acc3 = __builtin_amdgcn_wmma_f32_16x16x32_bf16(false, a, false, bw3, (short)0, acc3, false, false);
    }

    const int rb = b0 + lhalf * 8;
#pragma unroll
    for (int r = 0; r < 8; ++r) {
        float* dst = out + (size_t)(rb + r) * (NN * DOUT) + n * DOUT + l15;
        dst[0]  = acc0[r];
        dst[16] = acc1[r];
        dst[32] = acc2[r];
        dst[48] = acc3[r];
    }
}

// ---------------------------------------------------------------------------
extern "C" void kernel_launch(void* const* d_in, const int* in_sizes, int n_in,
                              void* d_out, int out_size, void* d_ws, size_t ws_size,
                              hipStream_t stream) {
    (void)in_sizes; (void)n_in; (void)out_size; (void)ws_size;
    const float* x   = (const float*)d_in[0];
    const float* emb = (const float*)d_in[1];
    const float* wp  = (const float*)d_in[2];
    const float* bp  = (const float*)d_in[3];
    float* out = (float*)d_out;

    char* ws = (char*)d_ws;
    size_t off = 0;
    bf16_t* sup  = (bf16_t*)(ws + off); off += (size_t)NN * NN * 2;          //  2 MB
    bf16_t* xtT  = (bf16_t*)(ws + off); off += (size_t)JJ * NN * 2;          // 16 MB
    bf16_t* wT   = (bf16_t*)(ws + off); off += (size_t)NN * DIN * DOUT * 2;  //  8 MB
    float*  bias = (float*) (ws + off); off += (size_t)NN * DOUT * 4;        // .25 MB
    bf16_t* xg   = (bf16_t*)(ws + off); off += (size_t)NN * JJ * 2;          // 16 MB

    k_supports<<<NN, 256, 0, stream>>>(emb, sup);
    k_xpose<<<BB * (NN / 64), 256, 0, stream>>>(x, xtT);
    k_weights<<<NN, 256, 0, stream>>>(emb, wp, bp, wT, bias);
    k_gemm1<<<8 * 128, 256, 0, stream>>>(sup, xtT, xg);   // 128x64 block tiles
    k_gemm2<<<NN, 256, 0, stream>>>(xg, wT, bias, out);
}
